// TGAEncoder_18949395710252
// MI455X (gfx1250) — matrix operand order repaired and checked by reading, not compiled, not run on hardware
//
#include <hip/hip_runtime.h>
#include <hip/hip_bf16.h>

typedef __attribute__((ext_vector_type(16))) _Float16 v16h;
typedef __attribute__((ext_vector_type(8)))  float    v8f;
typedef __attribute__((ext_vector_type(4)))  float    v4f;

#define NN 50000
#define EE 800000
#define DD 128
#define CONC 256
#define KPOOL 25000

// ---------------------------------------------------------------------------
// Generic fills
// ---------------------------------------------------------------------------
__global__ void fill_f32(float* p, long long n, float v) {
    long long i = (long long)blockIdx.x * blockDim.x + threadIdx.x;
    if (i < n) p[i] = v;
}
__global__ void fill_u32(unsigned* p, long long n, unsigned v) {
    long long i = (long long)blockIdx.x * blockDim.x + threadIdx.x;
    if (i < n) p[i] = v;
}

// ---------------------------------------------------------------------------
// Weight convert + transpose: W[K][C] f32 -> Wt[C][K] f16
// ---------------------------------------------------------------------------
__global__ void cvt_wt(const float* __restrict__ W, _Float16* __restrict__ Wt,
                       int K, int C) {
    int i = blockIdx.x * blockDim.x + threadIdx.x;
    if (i >= K * C) return;
    int k = i / C, c = i % C;
    Wt[(size_t)c * K + k] = (_Float16)W[i];
}

// ---------------------------------------------------------------------------
// Persistent-wave WMMA GEMM: Y[N,C] = act(X[N,KT] @ W + bias), W transposed f16.
// Each wave owns one 16-column stripe: B fragments (KT/32 x v16h) + bias are
// hoisted into registers once; the wave then grid-strides over 16-row tiles
// doing A-load -> v_wmma_f32_16x16x32_f16 -> store.
// A fragment (16x32 f16, MxK): lane m = lane&15; lanes<16 hold K 0..15 contiguous,
// lanes>=16 hold K 16..31 contiguous.  B fragment (32x16 f16, KxN): lane n=lane&15;
// lanes<16 hold K 0..15 contiguous, lanes>=16 hold K 16..31 contiguous.
// C/D: VGPR g, lane l -> row g+(l>=16?8:0), col l&15.
// act: 0 = none, 1 = relu
// ---------------------------------------------------------------------------
template <int KT>
__global__ __launch_bounds__(256) void wmma_gemm_t(
        const float* __restrict__ X, const _Float16* __restrict__ Wt,
        const float* __restrict__ bias, float* __restrict__ Y,
        int N, int C, int act) {
    constexpr int KB = KT / 32;
    int wave = threadIdx.x >> 5;
    int lane = threadIdx.x & 31;
    int c0 = (blockIdx.y * 8 + wave) * 16;
    if (c0 >= C) return;

    bool hi = lane >= 16;
    int col = c0 + (lane & 15);
    int abase = hi ? 16 : 0;

    // Hoist B fragments for this column stripe into registers (L2-resident once).
    const _Float16* brow = Wt + (size_t)col * KT + (hi ? 16 : 0);
    v16h bfrag[KB];
#pragma unroll
    for (int kb = 0; kb < KB; ++kb) bfrag[kb] = *(const v16h*)(brow + kb * 32);
    float bs = bias ? bias[col] : 0.0f;

    int nTiles = N / 16;
    for (int t = blockIdx.x; t < nTiles; t += gridDim.x) {
        int m = t * 16 + (lane & 15);
        const float* xrow = X + (size_t)m * KT;

        // Prefetch next tile's A rows into cache while this tile computes.
        int tn = t + gridDim.x;
        if (tn < nTiles)
            __builtin_prefetch(X + (size_t)(tn * 16 + (lane & 15)) * KT, 0, 0);

        v8f acc = {};
#pragma unroll
        for (int kb = 0; kb < KB; ++kb) {
            v16h a;
            v4f a0 = *(const v4f*)(xrow + kb * 32 + abase);
            v4f a1 = *(const v4f*)(xrow + kb * 32 + abase + 4);
            v4f a2 = *(const v4f*)(xrow + kb * 32 + abase + 8);
            v4f a3 = *(const v4f*)(xrow + kb * 32 + abase + 12);
#pragma unroll
            for (int j = 0; j < 4; ++j) {
                a[j]      = (_Float16)a0[j];
                a[4 + j]  = (_Float16)a1[j];
                a[8 + j]  = (_Float16)a2[j];
                a[12 + j] = (_Float16)a3[j];
            }
            acc = __builtin_amdgcn_wmma_f32_16x16x32_f16(
                false, a, false, bfrag[kb], (short)0, acc, false, false);
        }

#pragma unroll
        for (int g = 0; g < 8; ++g) {
            int row = t * 16 + g + (hi ? 8 : 0);
            float v = acc[g] + bs;
            if (act == 1) v = fmaxf(v, 0.0f);
            Y[(size_t)row * C + col] = v;
        }
    }
}

// ---------------------------------------------------------------------------
// GAT pieces
// ---------------------------------------------------------------------------
__global__ void gat_elr(const float* __restrict__ f, const float* __restrict__ al,
                        const float* __restrict__ ar, float* el, float* er, int N) {
    int i = blockIdx.x * blockDim.x + threadIdx.x;   // (n, h)
    if (i >= N * 4) return;
    int n = i >> 2, h = i & 3;
    const float* fp = f + (size_t)n * DD + h * 32;
    const float* ap = al + h * 32;
    const float* bp = ar + h * 32;
    float sl = 0.f, sr = 0.f;
#pragma unroll 8
    for (int d = 0; d < 32; ++d) { sl += fp[d] * ap[d]; sr += fp[d] * bp[d]; }
    el[i] = sl; er[i] = sr;
}

__device__ inline float leaky02(float v) { return v > 0.f ? v : 0.2f * v; }
__device__ inline unsigned f2key(float v) {
    unsigned b = __float_as_uint(v);
    return (b & 0x80000000u) ? ~b : (b | 0x80000000u);
}
__device__ inline float key2f(unsigned u) {
    unsigned b = (u & 0x80000000u) ? (u & 0x7FFFFFFFu) : ~u;
    return __uint_as_float(b);
}

__global__ void gat_edge_max(const float* __restrict__ el, const float* __restrict__ er,
                             const int* __restrict__ src, const int* __restrict__ dst,
                             unsigned* mu, int E) {
    int i = blockIdx.x * blockDim.x + threadIdx.x;   // (e, h)
    if (i >= E * 4) return;
    int e = i >> 2, h = i & 3;
    float v = leaky02(el[src[e] * 4 + h] + er[dst[e] * 4 + h]);
    atomicMax(&mu[dst[e] * 4 + h], f2key(v));
}

__global__ void gat_edge_exp(const float* __restrict__ el, const float* __restrict__ er,
                             const int* __restrict__ src, const int* __restrict__ dst,
                             const unsigned* __restrict__ mu,
                             float* __restrict__ ex, float* sbuf, int E) {
    int i = blockIdx.x * blockDim.x + threadIdx.x;
    if (i >= E * 4) return;
    int e = i >> 2, h = i & 3;
    float v = leaky02(el[src[e] * 4 + h] + er[dst[e] * 4 + h]);
    float x = __expf(v - key2f(mu[dst[e] * 4 + h]));
    ex[i] = x;
    atomicAdd(&sbuf[dst[e] * 4 + h], x);
}

__global__ void gat_edge_agg(const float* __restrict__ f,
                             const int* __restrict__ src, const int* __restrict__ dst,
                             const float* __restrict__ ex, const float* __restrict__ sbuf,
                             float* out, long long total) {
    long long i = (long long)blockIdx.x * blockDim.x + threadIdx.x;
    if (i >= total) return;
    int c = (int)(i & 127);
    long long e = i >> 7;
    int h = c >> 5;
    int d = dst[e], s = src[e];
    float a = ex[e * 4 + h] / sbuf[d * 4 + h];
    atomicAdd(&out[(size_t)d * DD + c], a * f[(size_t)s * DD + c]);
}

__global__ void gat_final(float* out, const float* __restrict__ res,
                          const float* __restrict__ b, int N) {
    long long i = (long long)blockIdx.x * blockDim.x + threadIdx.x;
    if (i >= (long long)N * DD) return;
    float v = out[i] + (res ? res[i] : 0.0f) + b[i & 127];
    out[i] = v > 0.f ? v : (__expf(v) - 1.0f);     // elu
}

// ---------------------------------------------------------------------------
// GCT: per-row max/mean, comb = (sigmoid(max)+sigmoid(mean)) * t  (in place)
// ---------------------------------------------------------------------------
__global__ void gct_comb(float* t, int N) {
    int n = blockIdx.x * blockDim.x + threadIdx.x;
    if (n >= N) return;
    float* row = t + (size_t)n * DD;
    float mx = -3.4e38f, sm = 0.f;
    for (int c = 0; c < DD; ++c) { float v = row[c]; mx = fmaxf(mx, v); sm += v; }
    float g = 1.f / (1.f + __expf(-mx)) + 1.f / (1.f + __expf(-sm * (1.0f / DD)));
    for (int c = 0; c < DD; ++c) row[c] *= g;
}

// ---------------------------------------------------------------------------
// Fusion / attention elementwise
// ---------------------------------------------------------------------------
__global__ void build_prod(const float* __restrict__ s13, const float* __restrict__ s31,
                           const float* __restrict__ cp, float* prod, int N) {
    long long i = (long long)blockIdx.x * blockDim.x + threadIdx.x;
    if (i >= (long long)N * CONC) return;
    int n = (int)(i >> 8), c = (int)(i & 255);
    float a, b;
    if (c < DD) { a = s13[(size_t)n * DD + c]; b = s31[(size_t)n * DD + c]; }
    else        { a = cp[(size_t)n * DD + (c - DD)]; b = a; }
    prod[i] = a * b;
}

__global__ void build_con(const float* __restrict__ sp, const float* __restrict__ ch,
                          float* con, int N) {
    long long i = (long long)blockIdx.x * blockDim.x + threadIdx.x;
    if (i >= (long long)N * CONC) return;
    int n = (int)(i >> 8), c = (int)(i & 255);
    con[i] = (c < DD) ? sp[(size_t)n * DD + c] : ch[(size_t)n * DD + (c - DD)];
}

__global__ void mul_inplace(float* a, const float* __restrict__ b, long long n) {
    long long i = (long long)blockIdx.x * blockDim.x + threadIdx.x;
    if (i < n) a[i] *= b[i];
}

__global__ void scores_keys(const float* __restrict__ attf, const float* __restrict__ pw,
                            const float* __restrict__ pb, float* scores, unsigned* keys, int N) {
    int n = blockIdx.x * blockDim.x + threadIdx.x;
    if (n >= N) return;
    const float* row = attf + (size_t)n * CONC;
    float s = pb[0];
    for (int c = 0; c < CONC; ++c) s += row[c] * pw[c];
    scores[n] = s;
    keys[n] = f2key(s);
}

// ---------------------------------------------------------------------------
// Top-K by exhaustive rank (descending, ties -> lower index first, matching
// jax.lax.top_k).  N=50000 -> 2.5G compares, LDS-tiled.
// ---------------------------------------------------------------------------
__global__ void topk_rank(const unsigned* __restrict__ keys, float* idxF, int* idxI,
                          int* sel, int N, int K) {
    int i = blockIdx.x * 256 + threadIdx.x;
    unsigned ki = (i < N) ? keys[i] : 0u;
    int rank = 0;
    __shared__ unsigned sk[256];
    for (int base = 0; base < N; base += 256) {
        int j = base + threadIdx.x;
        sk[threadIdx.x] = (j < N) ? keys[j] : 0u;
        __syncthreads();
        int lim = (N - base < 256) ? (N - base) : 256;
        for (int jj = 0; jj < lim; ++jj) {
            unsigned kj = sk[jj];
            int j2 = base + jj;
            if (kj > ki || (kj == ki && j2 < i)) rank++;
        }
        __syncthreads();
    }
    if (i < N && rank < K) {
        idxF[rank] = (float)i;
        idxI[rank] = i;
        sel[i] = 1;
    }
}

__global__ void gather_pooled(const float* __restrict__ attf, const int* __restrict__ idx,
                              float* pooled, int K) {
    long long i = (long long)blockIdx.x * blockDim.x + threadIdx.x;
    if (i >= (long long)K * CONC) return;
    int k = (int)(i >> 8), c = (int)(i & 255);
    pooled[i] = attf[(size_t)idx[k] * CONC + c];
}

__global__ void edge_mask_k(const int* __restrict__ sel, const int* __restrict__ src,
                            const int* __restrict__ dst, float* om, int E) {
    int e = blockIdx.x * blockDim.x + threadIdx.x;
    if (e >= E) return;
    om[e] = (sel[src[e]] && sel[dst[e]]) ? 1.0f : 0.0f;
}

// ---------------------------------------------------------------------------
// Host orchestration
// ---------------------------------------------------------------------------
extern "C" void kernel_launch(void* const* d_in, const int* in_sizes, int n_in,
                              void* d_out, int out_size, void* d_ws, size_t ws_size,
                              hipStream_t stream) {
    const int N = NN, E = EE;

    const float* feat    = (const float*)d_in[0];
    const float* gat1_W  = (const float*)d_in[1];
    const float* gat1_al = (const float*)d_in[2];
    const float* gat1_ar = (const float*)d_in[3];
    const float* gat1_b  = (const float*)d_in[4];
    const float* gat2_W  = (const float*)d_in[5];
    const float* gat2_al = (const float*)d_in[6];
    const float* gat2_ar = (const float*)d_in[7];
    const float* gat2_b  = (const float*)d_in[8];
    const float* gct_w1  = (const float*)d_in[9];
    const float* gct_b1  = (const float*)d_in[10];
    const float* gct_w2  = (const float*)d_in[11];
    const float* gct_b2  = (const float*)d_in[12];
    const float* gct_w3  = (const float*)d_in[13];
    const float* gct_b3  = (const float*)d_in[14];
    const float* gct_w4  = (const float*)d_in[15];
    const float* gct_b4  = (const float*)d_in[16];
    const float* ff13_w  = (const float*)d_in[17];
    const float* ff13_b  = (const float*)d_in[18];
    const float* ff31_w  = (const float*)d_in[19];
    const float* ff31_b  = (const float*)d_in[20];
    const float* ffmc_w1 = (const float*)d_in[21];
    const float* ffmc_b1 = (const float*)d_in[22];
    const float* ffmc_w2 = (const float*)d_in[23];
    const float* ffmc_b2 = (const float*)d_in[24];
    const float* ffu_w1  = (const float*)d_in[25];
    const float* ffu_b1  = (const float*)d_in[26];
    const float* ffu_w2  = (const float*)d_in[27];
    const float* ffu_b2  = (const float*)d_in[28];
    const float* att_w1  = (const float*)d_in[29];
    const float* att_b1  = (const float*)d_in[30];
    const float* att_w2  = (const float*)d_in[31];
    const float* att_b2  = (const float*)d_in[32];
    const float* pool_w  = (const float*)d_in[33];
    const float* pool_b  = (const float*)d_in[34];
    const int*   src     = (const int*)d_in[35];
    const int*   dst     = (const int*)d_in[36];

    float* out = (float*)d_out;
    float* outPooled = out;                                  // K*256
    float* outFused  = out + (size_t)KPOOL * CONC;           // N*128
    float* outIdxF   = outFused + (size_t)N * DD;            // K
    float* outMask   = outIdxF + KPOOL;                      // E

    // Bump allocator over workspace
    char* wsb = (char*)d_ws;
    size_t off = 0;
    auto alloc = [&](size_t bytes) -> void* {
        void* p = wsb + off;
        off += (bytes + 255) & ~(size_t)255;
        return p;
    };

    // f16 transposed weights
    _Float16* wtGat1  = (_Float16*)alloc(128 * 128 * 2);
    _Float16* wtGat2  = (_Float16*)alloc(128 * 128 * 2);
    _Float16* wtGct1  = (_Float16*)alloc(128 * 128 * 2);
    _Float16* wtGct2  = (_Float16*)alloc(128 * 128 * 2);
    _Float16* wtGct3  = (_Float16*)alloc(128 * 128 * 2);
    _Float16* wtGct4  = (_Float16*)alloc(128 * 128 * 2);
    _Float16* wtFf13  = (_Float16*)alloc(128 * 128 * 2);
    _Float16* wtFf31  = (_Float16*)alloc(128 * 128 * 2);
    _Float16* wtFfmc1 = (_Float16*)alloc(128 * 128 * 2);
    _Float16* wtFfmc2 = (_Float16*)alloc(128 * 128 * 2);
    _Float16* wtFfu1  = (_Float16*)alloc(256 * 128 * 2);
    _Float16* wtFfu2  = (_Float16*)alloc(128 * 128 * 2);
    _Float16* wtAtt1  = (_Float16*)alloc(256 * 256 * 2);
    _Float16* wtAtt2  = (_Float16*)alloc(256 * 256 * 2);

    // activations
    float* fbuf    = (float*)alloc((size_t)N * DD * 4);
    float* h1      = (float*)alloc((size_t)N * DD * 4);
    float* spatial = (float*)alloc((size_t)N * DD * 4);
    float* tbuf    = (float*)alloc((size_t)N * DD * 4);
    float* channel = (float*)alloc((size_t)N * DD * 4);
    float* cp      = (float*)alloc((size_t)N * DD * 4);
    float* s13     = (float*)alloc((size_t)N * DD * 4);
    float* s31     = (float*)alloc((size_t)N * DD * 4);
    float* tmp128  = (float*)alloc((size_t)N * DD * 4);
    float* prodb   = (float*)alloc((size_t)N * CONC * 4);    // also reused as tmp256
    float* conb    = (float*)alloc((size_t)N * CONC * 4);
    float* attb    = (float*)alloc((size_t)N * CONC * 4);

    float*    elb  = (float*)alloc((size_t)N * 4 * 4);
    float*    erb  = (float*)alloc((size_t)N * 4 * 4);
    unsigned* mu   = (unsigned*)alloc((size_t)N * 4 * 4);
    float*    sbuf = (float*)alloc((size_t)N * 4 * 4);
    float*    exb  = (float*)alloc((size_t)E * 4 * 4);
    float*    scr  = (float*)alloc((size_t)N * 4);
    unsigned* keys = (unsigned*)alloc((size_t)N * 4);
    int*      idxI = (int*)alloc((size_t)KPOOL * 4);
    int*      sel  = (int*)alloc((size_t)N * 4);
    (void)ws_size; (void)n_in; (void)in_sizes; (void)out_size;

    auto gblocks = [](long long n) { return (unsigned)((n + 255) / 256); };

    // convert weights
    auto cvt = [&](const float* W, _Float16* Wt, int K, int C) {
        cvt_wt<<<gblocks((long long)K * C), 256, 0, stream>>>(W, Wt, K, C);
    };
    cvt(gat1_W, wtGat1, 128, 128);  cvt(gat2_W, wtGat2, 128, 128);
    cvt(gct_w1, wtGct1, 128, 128);  cvt(gct_w2, wtGct2, 128, 128);
    cvt(gct_w3, wtGct3, 128, 128);  cvt(gct_w4, wtGct4, 128, 128);
    cvt(ff13_w, wtFf13, 128, 128);  cvt(ff31_w, wtFf31, 128, 128);
    cvt(ffmc_w1, wtFfmc1, 128, 128); cvt(ffmc_w2, wtFfmc2, 128, 128);
    cvt(ffu_w1, wtFfu1, 256, 128);  cvt(ffu_w2, wtFfu2, 128, 128);
    cvt(att_w1, wtAtt1, 256, 256);  cvt(att_w2, wtAtt2, 256, 256);

    // Persistent-wave GEMM dispatch: enough blocks to fill the device, each
    // wave keeps its B stripe in registers and strides over row tiles.
    const unsigned GEMM_BLOCKS = 1024;
    auto gemm = [&](const float* X, const _Float16* Wt, const float* bias, float* Y,
                    int K, int C, int act) {
        dim3 grid(GEMM_BLOCKS, C / 128);
        if (K == 128)
            wmma_gemm_t<128><<<grid, 256, 0, stream>>>(X, Wt, bias, Y, N, C, act);
        else
            wmma_gemm_t<256><<<grid, 256, 0, stream>>>(X, Wt, bias, Y, N, C, act);
    };

    // One GAT layer: dest = elu(segsoftmax-agg(X@W) + res + bias)
    auto gat = [&](const float* Xin, const _Float16* Wt, const float* al, const float* ar,
                   const float* bias, const float* res, float* dest) {
        gemm(Xin, Wt, nullptr, fbuf, 128, 128, 0);
        gat_elr<<<gblocks((long long)N * 4), 256, 0, stream>>>(fbuf, al, ar, elb, erb, N);
        fill_u32<<<gblocks((long long)N * 4), 256, 0, stream>>>(mu, (long long)N * 4, 0u);
        fill_f32<<<gblocks((long long)N * 4), 256, 0, stream>>>(sbuf, (long long)N * 4, 0.f);
        fill_f32<<<gblocks((long long)N * DD), 256, 0, stream>>>(dest, (long long)N * DD, 0.f);
        gat_edge_max<<<gblocks((long long)E * 4), 256, 0, stream>>>(elb, erb, src, dst, mu, E);
        gat_edge_exp<<<gblocks((long long)E * 4), 256, 0, stream>>>(elb, erb, src, dst, mu, exb, sbuf, E);
        gat_edge_agg<<<gblocks((long long)E * DD), 256, 0, stream>>>(fbuf, src, dst, exb, sbuf, dest, (long long)E * DD);
        gat_final<<<gblocks((long long)N * DD), 256, 0, stream>>>(dest, res, bias, N);
    };

    // ---- spatial branch: 2-layer GAT ----
    gat(feat, wtGat1, gat1_al, gat1_ar, gat1_b, nullptr, h1);
    gat(h1,   wtGat2, gat2_al, gat2_ar, gat2_b, h1, spatial);

    // ---- channel branch: GCT ----
    gemm(feat,   wtGct1, gct_b1, tmp128, 128, 128, 1);
    gemm(tmp128, wtGct2, gct_b2, tbuf,   128, 128, 0);
    gct_comb<<<gblocks(N), 256, 0, stream>>>(tbuf, N);
    gemm(tbuf,   wtGct3, gct_b3, tmp128, 128, 128, 1);
    gemm(tmp128, wtGct4, gct_b4, channel, 128, 128, 0);

    // ---- feature fusion ----
    gemm(spatial, wtFf13, ff13_b, s13, 128, 128, 0);
    gemm(spatial, wtFf31, ff31_b, s31, 128, 128, 0);
    gemm(channel, wtFfmc1, ffmc_b1, tmp128, 128, 128, 1);
    gemm(tmp128,  wtFfmc2, ffmc_b2, cp, 128, 128, 0);
    build_prod<<<gblocks((long long)N * CONC), 256, 0, stream>>>(s13, s31, cp, prodb, N);
    gemm(prodb,  wtFfu1, ffu_b1, tmp128, 256, 128, 1);
    gemm(tmp128, wtFfu2, ffu_b2, outFused, 128, 128, 0);

    // ---- attention gating ----
    build_con<<<gblocks((long long)N * CONC), 256, 0, stream>>>(spatial, channel, conb, N);
    gemm(conb,  wtAtt1, att_b1, prodb, 256, 256, 1);   // prodb reused as tmp256
    gemm(prodb, wtAtt2, att_b2, attb, 256, 256, 0);
    mul_inplace<<<gblocks((long long)N * CONC), 256, 0, stream>>>(attb, conb, (long long)N * CONC);

    // ---- scores + top-k pooling ----
    scores_keys<<<gblocks(N), 256, 0, stream>>>(attb, pool_w, pool_b, scr, keys, N);
    fill_f32<<<gblocks(N), 256, 0, stream>>>((float*)sel, N, 0.f);  // sel=0 (bit pattern 0)
    topk_rank<<<(N + 255) / 256, 256, 0, stream>>>(keys, outIdxF, idxI, sel, N, KPOOL);
    gather_pooled<<<gblocks((long long)KPOOL * CONC), 256, 0, stream>>>(attb, idxI, outPooled, KPOOL);
    edge_mask_k<<<gblocks(E), 256, 0, stream>>>(sel, src, dst, outMask, E);
}
